// Attention_6468220748058
// MI455X (gfx1250) — compile-verified, matrix operand order checked by reading, not run on hardware
//
#include <hip/hip_runtime.h>
#include <hip/hip_bf16.h>
#include <math.h>

typedef __attribute__((ext_vector_type(16))) __bf16 v16bf;
typedef __attribute__((ext_vector_type(8)))  __bf16 v8bf;
typedef __attribute__((ext_vector_type(8)))  float  v8f;
typedef __attribute__((ext_vector_type(4)))  unsigned int u32x4;
typedef __attribute__((ext_vector_type(8)))  int          i32x8;
typedef __attribute__((ext_vector_type(4)))  int          i32x4;

#define B_ 2
#define S_ 2048
#define E_ 1024
#define H_ 16
#define D_ 64

#if defined(__gfx1250__) && __has_builtin(__builtin_amdgcn_tensor_load_to_lds) && \
    __has_builtin(__builtin_amdgcn_s_wait_tensorcnt)
#define ATH_TDM 1
#else
#define ATH_TDM 0
#endif

// ---------------------------------------------------------------------------
// Fragment loaders
// ---------------------------------------------------------------------------
// bf16 row fragment: two contiguous 8-elem chunks at k = hi*8 and 16+hi*8
// (CDNA5 16-bit A/B 16x32 VGPR layout), each a 16B-aligned load.
static __device__ __forceinline__ v16bf load_frag(const __bf16* row, int hi) {
    union { v16bf v; v8bf h[2]; } u;
    u.h[0] = *(const v8bf*)(row + hi * 8);
    u.h[1] = *(const v8bf*)(row + 16 + hi * 8);
    return u.v;
}

// fp32 LDS row -> bf16 fragment (convert on read; LDS tile staged by TDM).
static __device__ __forceinline__ v16bf cvt_frag(const float* row, int hi) {
    v16bf r;
#pragma unroll
    for (int i = 0; i < 8; ++i) r[i] = (__bf16)row[hi * 8 + i];
#pragma unroll
    for (int i = 0; i < 8; ++i) r[8 + i] = (__bf16)row[16 + hi * 8 + i];
    return r;
}

// ---------------------------------------------------------------------------
// Tensor Data Mover: 2D tile Global -> LDS.  Builds the D# per ISA ch.8:
//  group0: count=1 | lds_addr | global_addr[56:0] | type=2
//  group1: data_size | tensor_dim0/1 | tile_dim0/1 | tensor_dim0_stride
// data_size code: 0=1B 1=2B 2=4B 3=8B.  Dims/strides in elements.
// ---------------------------------------------------------------------------
#if ATH_TDM
static __device__ __forceinline__ void tdm_load_2d(
    unsigned lds_off, const void* gptr, unsigned ds_code,
    unsigned tile_d0, unsigned tile_d1,
    unsigned long long tens_d0, unsigned long long tens_d1,
    unsigned long long stride0) {
    unsigned long long ga = (unsigned long long)gptr;
    u32x4 g0;
    g0[0] = 1u;                                            // count=1, user desc
    g0[1] = lds_off;                                       // LDS byte address
    g0[2] = (unsigned)(ga & 0xffffffffu);
    g0[3] = (unsigned)((ga >> 32) & 0x1ffffffu) | (2u << 30);  // addr[56:32] | type=2
    i32x8 g1;
    g1[0] = (int)(ds_code << 16);                          // wg_mask=0, data_size
    g1[1] = (int)((tens_d0 & 0xffffu) << 16);              // abar=0 | tensor_dim0.lo
    g1[2] = (int)(((tens_d0 >> 16) & 0xffffu) | ((tens_d1 & 0xffffu) << 16));
    g1[3] = (int)(((tens_d1 >> 16) & 0xffffu) | (tile_d0 << 16));
    g1[4] = (int)(tile_d1 & 0xffffu);                      // tile_dim1 | tile_dim2=0
    g1[5] = (int)(stride0 & 0xffffffffu);                  // tensor_dim0_stride.lo
    g1[6] = (int)((stride0 >> 32) & 0xffffu);              // .hi | dim1_stride=0
    g1[7] = 0;
    const i32x4 z4 = {0, 0, 0, 0};                         // 2D: groups 2/3 unused
    const i32x8 z8 = {0, 0, 0, 0, 0, 0, 0, 0};
    __builtin_amdgcn_tensor_load_to_lds(g0, g1, z4, z4, z8, 0);
}
#endif

static __device__ __forceinline__ unsigned lds_off_of(const void* p) {
    return (unsigned)(unsigned long long)p;                // LDS aperture: addr[31:0]
}

// ---------------------------------------------------------------------------
// Kernel 1: projection  out = x @ W^T  (templated on V-transpose layout).
// Block 256 (8 waves), tile 128(M) x 64(N), K-step 32.  x/W tiles staged fp32
// into LDS by TDM; bf16 conversion happens on the fragment read.
// ---------------------------------------------------------------------------
template <bool VT>
__global__ __launch_bounds__(256)
void proj_kernel(const float* __restrict__ x,
                 const float* __restrict__ W,
                 __bf16* __restrict__ O) {
    __shared__ __align__(16) float la[128][32];   // 16 KB
    __shared__ __align__(16) float lb[64][32];    //  8 KB

    const int t   = threadIdx.x;
    const int bm0 = blockIdx.x * 128;
    const int bn0 = blockIdx.y * 64;

    const int lane = t & 31, wave = t >> 5;
    const int wm = (wave & 3) * 32;
    const int wn = (wave >> 2) * 32;
    const int lm = lane & 15, hi = lane >> 4;

    v8f acc[2][2] = {};

    for (int kc = 0; kc < E_; kc += 32) {
        __syncthreads();                          // previous tile consumed
#if ATH_TDM
        if (t == 0) {
            tdm_load_2d(lds_off_of(&la[0][0]), x + (size_t)bm0 * E_ + kc, 2,
                        32, 128, E_, (unsigned long long)B_ * S_, E_);
            tdm_load_2d(lds_off_of(&lb[0][0]), W + (size_t)bn0 * E_ + kc, 2,
                        32, 64, E_, E_, E_);
        }
        __builtin_amdgcn_s_wait_tensorcnt(0);
#else
        for (int i = t; i < 128 * 32; i += 256)
            (&la[0][0])[i] = x[((size_t)bm0 + (i >> 5)) * E_ + kc + (i & 31)];
        for (int i = t; i < 64 * 32; i += 256)
            (&lb[0][0])[i] = W[((size_t)bn0 + (i >> 5)) * E_ + kc + (i & 31)];
#endif
        __syncthreads();

        v16bf af[2], bfr[2];
#pragma unroll
        for (int i = 0; i < 2; ++i) af[i]  = cvt_frag(&la[wm + i * 16 + lm][0], hi);
#pragma unroll
        for (int j = 0; j < 2; ++j) bfr[j] = cvt_frag(&lb[wn + j * 16 + lm][0], hi);
#pragma unroll
        for (int i = 0; i < 2; ++i)
#pragma unroll
            for (int j = 0; j < 2; ++j)
                acc[i][j] = __builtin_amdgcn_wmma_f32_16x16x32_bf16(
                    false, af[i], false, bfr[j], (short)0, acc[i][j], false, false);
    }

    // Whole 128-row tile lies in one batch (2048 % 128 == 0): hoist b/s split.
    const int b  = bm0 >> 11;
    const int s0 = (bm0 & 2047) + wm + hi * 8;    // + i*16 + r
#pragma unroll
    for (int j = 0; j < 2; ++j) {
        const int n  = bn0 + wn + j * 16 + lm;
        const int h  = n >> 6, dd = n & 63;
        __bf16* p = VT ? (O + ((size_t)(b * H_ + h) * D_ + dd) * S_ + s0)
                       : (O + ((size_t)(b * H_ + h) * S_ + s0) * D_ + dd);
#pragma unroll
        for (int i = 0; i < 2; ++i)
#pragma unroll
            for (int r = 0; r < 8; ++r) {
                const int mrel = i * 16 + r;
                const __bf16 val = (__bf16)acc[i][j][r];
                if (VT) p[mrel] = val;
                else    p[(size_t)mrel * D_] = val;
            }
    }
}

// ---------------------------------------------------------------------------
// Kernel 2: flash attention, distance-weighted logits, online softmax.
// Grid (S/64, B*H); block 128 (4 waves, one 16-row q slice each).
// K (32x64) and V^T (64x32) tiles staged bf16 into LDS by TDM per iteration.
// ---------------------------------------------------------------------------
__global__ __launch_bounds__(128)
void flash_attn_kernel(const __bf16* __restrict__ q_ws,
                       const __bf16* __restrict__ k_ws,
                       const __bf16* __restrict__ v_ws,
                       float* __restrict__ o_ws) {
    __shared__ __align__(16) __bf16 k_tile[32][64];     // 4 KB, rows = s
    __shared__ __align__(16) __bf16 v_tile[64][32];     // 4 KB, rows = d
    __shared__ __align__(16) __bf16 p_lds[4][16][32];   // per-wave P relayout

    const int t = threadIdx.x, lane = t & 31, w = t >> 5;
    const int lm = lane & 15, hi = lane >> 4;
    const int bh = blockIdx.y;
    const int q0 = blockIdx.x * 64 + w * 16;

    const float scale = 0.03125f;                 // E^-0.5
    const float invS  = 1.0f / (float)S_;

    const __bf16* qrow = q_ws + ((size_t)bh * S_ + q0 + lm) * D_;
    const v16bf aq0 = load_frag(qrow, hi);
    const v16bf aq1 = load_frag(qrow + 32, hi);

    v8f o[4] = {};
    float l[8], mx[8];
#pragma unroll
    for (int r = 0; r < 8; ++r) { l[r] = 0.0f; mx[r] = -INFINITY; }

    const float qf = (float)(q0 + hi * 8);

    for (int kt = 0; kt < S_; kt += 32) {
        __syncthreads();                          // previous tiles consumed
#if ATH_TDM
        if (t == 0) {
            tdm_load_2d(lds_off_of(&k_tile[0][0]),
                        k_ws + ((size_t)bh * S_ + kt) * D_, 1,
                        D_, 32, D_, S_, D_);
            tdm_load_2d(lds_off_of(&v_tile[0][0]),
                        v_ws + (size_t)bh * D_ * S_ + kt, 1,
                        32, D_, S_, D_, S_);
        }
        __builtin_amdgcn_s_wait_tensorcnt(0);
#else
        for (int i = t; i < 32 * D_; i += 128)    // K tile is globally contiguous
            (&k_tile[0][0])[i] = k_ws[((size_t)bh * S_ + kt) * D_ + i];
        for (int i = t; i < D_ * 32; i += 128)
            (&v_tile[0][0])[i] = v_ws[((size_t)bh * D_ + (i >> 5)) * S_ + kt + (i & 31)];
#endif
        __builtin_prefetch(k_ws + ((size_t)bh * S_ + kt + 32) * D_, 0, 1);
        __syncthreads();

        // ---- S = Q K^T over a 16x32 score tile (two 16-col subtiles) ----
        v8f s0 = {}, s1 = {};
        {
            v16bf b00 = load_frag(&k_tile[lm][0],       hi);
            v16bf b01 = load_frag(&k_tile[lm][32],      hi);
            v16bf b10 = load_frag(&k_tile[16 + lm][0],  hi);
            v16bf b11 = load_frag(&k_tile[16 + lm][32], hi);
            s0 = __builtin_amdgcn_wmma_f32_16x16x32_bf16(false, aq0, false, b00, (short)0, s0, false, false);
            s0 = __builtin_amdgcn_wmma_f32_16x16x32_bf16(false, aq1, false, b01, (short)0, s0, false, false);
            s1 = __builtin_amdgcn_wmma_f32_16x16x32_bf16(false, aq0, false, b10, (short)0, s1, false, false);
            s1 = __builtin_amdgcn_wmma_f32_16x16x32_bf16(false, aq1, false, b11, (short)0, s1, false, false);
        }

        const float k0f = (float)(kt + lm);
        const float k1f = (float)(kt + 16 + lm);

        // ---- distance weighting + online softmax (rows in 16-lane halves) --
#pragma unroll
        for (int r = 0; r < 8; ++r) {
            const float qr = qf + (float)r;
            float v0 = s0[r] * scale * fabsf(qr - k0f) * invS;
            float v1 = s1[r] * scale * fabsf(qr - k1f) * invS;
            float rmax = fmaxf(v0, v1);
#pragma unroll
            for (int off = 1; off < 16; off <<= 1)
                rmax = fmaxf(rmax, __shfl_xor(rmax, off, 32));
            const float mnew = fmaxf(mx[r], rmax);
            const float corr = __expf(mx[r] - mnew);
            const float p0 = __expf(v0 - mnew);
            const float p1 = __expf(v1 - mnew);
            float rsum = p0 + p1;
#pragma unroll
            for (int off = 1; off < 16; off <<= 1)
                rsum += __shfl_xor(rsum, off, 32);
            l[r]  = l[r] * corr + rsum;
            mx[r] = mnew;
#pragma unroll
            for (int d = 0; d < 4; ++d) o[d][r] *= corr;
            const int mrow = r + hi * 8;
            p_lds[w][mrow][lm]      = (__bf16)p0;
            p_lds[w][mrow][16 + lm] = (__bf16)p1;
        }

        // ---- O += P V ----
        const v16bf ap = load_frag(&p_lds[w][lm][0], hi);
#pragma unroll
        for (int d = 0; d < 4; ++d) {
            v16bf bv = load_frag(&v_tile[d * 16 + lm][0], hi);
            o[d] = __builtin_amdgcn_wmma_f32_16x16x32_bf16(false, ap, false, bv, (short)0, o[d], false, false);
        }
    }

    // ---- normalize and scatter to [b][s][e] fp32 ----
    const int b = bh >> 4, h = bh & 15;
#pragma unroll
    for (int r = 0; r < 8; ++r) {
        const float inv = 1.0f / l[r];
        const int q = q0 + r + hi * 8;
        float* op = o_ws + ((size_t)b * S_ + q) * E_ + h * D_;
#pragma unroll
        for (int d = 0; d < 4; ++d)
            op[d * 16 + lm] = o[d][r] * inv;
    }
}

// ---------------------------------------------------------------------------
// Kernel 3: LayerNorm over E=1024, one block (256 threads) per row.
// ---------------------------------------------------------------------------
__global__ __launch_bounds__(256)
void layernorm_kernel(const float* __restrict__ o_ws,
                      const float* __restrict__ gamma,
                      const float* __restrict__ beta,
                      float* __restrict__ out) {
    __shared__ float red[16];
    const int row = blockIdx.x;
    const int t = threadIdx.x;

    const float4 v = *(const float4*)(o_ws + (size_t)row * E_ + t * 4);
    float s  = v.x + v.y + v.z + v.w;
    float s2 = v.x * v.x + v.y * v.y + v.z * v.z + v.w * v.w;
#pragma unroll
    for (int off = 1; off < 32; off <<= 1) {
        s  += __shfl_xor(s,  off, 32);
        s2 += __shfl_xor(s2, off, 32);
    }
    const int wv = t >> 5, ln = t & 31;
    if (ln == 0) { red[wv] = s; red[8 + wv] = s2; }
    __syncthreads();
    float ts = 0.0f, ts2 = 0.0f;
#pragma unroll
    for (int i = 0; i < 8; ++i) { ts += red[i]; ts2 += red[8 + i]; }

    const float mu   = ts * (1.0f / (float)E_);
    const float var  = ts2 * (1.0f / (float)E_) - mu * mu;
    const float rstd = rsqrtf(var + 1e-5f);

    const float4 g  = *(const float4*)(gamma + t * 4);
    const float4 be = *(const float4*)(beta  + t * 4);
    float4 y;
    y.x = (v.x - mu) * rstd * g.x + be.x;
    y.y = (v.y - mu) * rstd * g.y + be.y;
    y.z = (v.z - mu) * rstd * g.z + be.z;
    y.w = (v.w - mu) * rstd * g.w + be.w;
    *(float4*)(out + (size_t)row * E_ + t * 4) = y;
}

// ---------------------------------------------------------------------------
extern "C" void kernel_launch(void* const* d_in, const int* in_sizes, int n_in,
                              void* d_out, int out_size, void* d_ws, size_t ws_size,
                              hipStream_t stream) {
    const float* x  = (const float*)d_in[0];
    const float* Wq = (const float*)d_in[1];
    const float* Wk = (const float*)d_in[2];
    const float* Wv = (const float*)d_in[3];
    const float* g  = (const float*)d_in[4];
    const float* be = (const float*)d_in[5];
    float* out = (float*)d_out;

    char* ws = (char*)d_ws;
    const size_t qkv_bytes = (size_t)B_ * H_ * S_ * D_ * sizeof(__bf16);  // 8 MB each
    __bf16* q_ws = (__bf16*)(ws);
    __bf16* k_ws = (__bf16*)(ws + qkv_bytes);
    __bf16* v_ws = (__bf16*)(ws + 2 * qkv_bytes);
    float*  o_ws = (float*)(ws + 3 * qkv_bytes);                          // 16.8 MB

    const dim3 pg((B_ * S_) / 128, E_ / 64), pb(256);
    proj_kernel<false><<<pg, pb, 0, stream>>>(x, Wq, q_ws);
    proj_kernel<false><<<pg, pb, 0, stream>>>(x, Wk, k_ws);
    proj_kernel<true ><<<pg, pb, 0, stream>>>(x, Wv, v_ws);
    flash_attn_kernel<<<dim3(S_ / 64, B_ * H_), dim3(128), 0, stream>>>(
        q_ws, k_ws, v_ws, o_ws);
    layernorm_kernel<<<dim3(B_ * S_), dim3(256), 0, stream>>>(o_ws, g, be, out);
}
